// STGCN_74320114090518
// MI455X (gfx1250) — matrix operand order repaired and checked by reading, not compiled, not run on hardware
//
#include <hip/hip_runtime.h>
#include <hip/hip_bf16.h>

// ---------------------------------------------------------------------------
// STGCN forward for MI455X (gfx1250, wave32, WMMA).
// Bandwidth-bound (~290 GF vs <1 GB HBM traffic @ 23.3 TB/s):
//  * all intermediates f16, padded layouts -> branch-free b128 fragment fills
//  * densified 224x224 f16 L-hat; its z operand staged transposed in LDS so
//    B-fragments are ds_load_b128
//  * M-blocked GEMM waves reuse B fragments across MT accumulators
//  * BatchNorm folded algebraically into the next linear consumer:
//    sum_k w[k](x[k]*sc+of) = sc*conv(x) + of*sum_k w[k]  -> the in-place
//    normalize pass (~190 MB of HBM traffic) disappears; only a vectorized
//    stats pass remains.
// Zero-padding invariants: weight K-pads and L-hat node-pads are 0, so any
// garbage from padded/overshoot activation regions contributes a*0 = 0.
// ---------------------------------------------------------------------------

typedef __attribute__((ext_vector_type(16))) _Float16 v16h;
typedef __attribute__((ext_vector_type(8)))  _Float16 v8h;
typedef __attribute__((ext_vector_type(8)))  float    v8f;

#define NN   210        // real node count
#define NNP  224        // padded node stride (14 * 16)
#define NE   4200       // edge count
#define ZSTR 232        // LDS transpose row stride in halves (224 + 8)

__device__ __forceinline__ v8f wmma_f16(const v16h& a, const v16h& b, const v8f& c) {
    return __builtin_amdgcn_wmma_f32_16x16x32_f16(false, a, false, b, (short)0, c,
                                                  false, false);
}
__device__ __forceinline__ v16h cat8(v8h a, v8h b) {
    return __builtin_shufflevector(a, b, 0, 1, 2, 3, 4, 5, 6, 7,
                                         8, 9, 10, 11, 12, 13, 14, 15);
}

// ------------------------- graph preprocessing -----------------------------

__global__ void k_zero(float* p, int n) {
    int i = blockIdx.x * blockDim.x + threadIdx.x;
    if (i < n) p[i] = 0.f;
}

__global__ void k_deg(const int* __restrict__ ei, const float* __restrict__ ea,
                      float* __restrict__ deg) {
    int e = blockIdx.x * blockDim.x + threadIdx.x;
    if (e < NE) atomicAdd(&deg[ei[e]], ea[e]);          // row = ei[0][e]
}

__global__ void k_build_L(const int* __restrict__ ei, const float* __restrict__ ea,
                          const float* __restrict__ deg, float* __restrict__ L) {
    int e = blockIdx.x * blockDim.x + threadIdx.x;
    if (e < NE) {
        int r = ei[e], c = ei[NE + e];
        float dr = deg[r] > 0.f ? rsqrtf(deg[r]) : 0.f;
        float dc = deg[c] > 0.f ? rsqrtf(deg[c]) : 0.f;
        atomicAdd(&L[r * NN + c], -dr * ea[e] * dc);
    }
}

__global__ void k_pack_L(const float* __restrict__ L, _Float16* __restrict__ Lh) {
    int i = blockIdx.x * blockDim.x + threadIdx.x;
    if (i < NNP * NNP) {
        int k = i % NNP, r = i / NNP;
        Lh[i] = (_Float16)((r < NN && k < NN) ? L[r * NN + k] : 0.f);
    }
}

// x (32,64,210,14) f32 -> xp (32,64,224,16) f16 zero padded
__global__ void k_pack_x(const float* __restrict__ x, _Float16* __restrict__ xp,
                         int total) {
    int i = blockIdx.x * blockDim.x + threadIdx.x;
    if (i >= total) return;
    int ci = i & 15;  int rest = i >> 4;
    int node = rest % NNP;  int bt = rest / NNP;
    float v = 0.f;
    if (node < NN && ci < 14) v = x[((size_t)bt * NN + node) * 14 + ci];
    xp[i] = (_Float16)v;
}

// w (3,KT,CIN,COUT) f32 -> wp (3,COUT,KPAD) f16, K = dt*CINP+ci, zero padded
__global__ void k_pack_wt(const float* __restrict__ w, _Float16* __restrict__ wp,
                          int KT, int CIN, int CINP, int COUT, int KPAD) {
    int i = blockIdx.x * blockDim.x + threadIdx.x;
    if (i >= 3 * COUT * KPAD) return;
    int k = i % KPAD;  int rest = i / KPAD;
    int co = rest % COUT;  int g = rest / COUT;
    int dt = k / CINP, ci = k - dt * CINP;
    float v = 0.f;
    if (dt < KT && ci < CIN) v = w[(((size_t)g * KT + dt) * CIN + ci) * COUT + co];
    wp[i] = (_Float16)v;
}

// per-(gate,co) weight tap sums: wsum[g*COUT+co] = sum_k wp[(g*COUT+co)*KPAD+k]
__global__ void k_wsum(const _Float16* __restrict__ wp, float* __restrict__ wsum,
                       int COUT, int KPAD) {
    int i = blockIdx.x * blockDim.x + threadIdx.x;
    if (i < 3 * COUT) {
        const _Float16* p = wp + (size_t)i * KPAD;
        float s = 0.f;
        for (int k = 0; k < KPAD; ++k) s += (float)p[k];
        wsum[i] = s;
    }
}

// W (3,64,64)[h][g] f32 -> Wt (3,64,64)[g][h] f16
__global__ void k_pack_cw(const float* __restrict__ W, _Float16* __restrict__ Wt) {
    int i = blockIdx.x * blockDim.x + threadIdx.x;
    if (i < 3 * 64 * 64) {
        int h = i % 64;  int rest = i / 64;
        int g = rest % 64;  int s = rest / 64;
        Wt[i] = (_Float16)W[((size_t)s * 64 + h) * 64 + g];
    }
}

// --------------- gated temporal conv: relu(P * sigmoid(Q) + R) -------------
// x: (B,Tin,NNP,CINP) f16  wp: (3,COUT,KPAD) f16  out: (B,Tout,NNP,COUT) f16
// BNIN: input is pre-BN; apply y = x*sc[node]+of[node] algebraically in the
// epilogue via acc*sc + of*wsum.  MT row-tiles share B fragments per K-step.
template <int KPAD, int CINP, int COUT, int MT, bool BNIN>
__global__ __launch_bounds__(256)
void k_tconv(const _Float16* __restrict__ x, const _Float16* __restrict__ wp,
             const float* __restrict__ bias, const float* __restrict__ bnsc,
             const float* __restrict__ bnof, const float* __restrict__ wsum,
             _Float16* __restrict__ out, int B, int Tin, int Tout,
             int mChunks, int nTiles) {
    int wave = blockIdx.x * 8 + (threadIdx.x >> 5);
    if (wave >= nTiles) return;                         // wave-uniform
    constexpr int NT = COUT / 16;
    int wv = wave;
    int nIdx = wv % NT;      wv /= NT;
    int mc   = wv % mChunks; wv /= mChunks;
    int node = wv % NN;
    int b    = wv / NN;

    const int l    = threadIdx.x & 31;
    const int lo   = l & 15;
    const int hsel = l >> 4;
    const int co   = nIdx * 16 + lo;
    const int tb   = mc * (MT * 16);

    const _Float16* w0 = wp + (size_t)co * KPAD;
    const _Float16* w1 = w0 + (size_t)COUT * KPAD;
    const _Float16* w2 = w1 + (size_t)COUT * KPAD;

    size_t abase[MT];
#pragma unroll
    for (int mi = 0; mi < MT; ++mi)
        abase[mi] = ((size_t)(b * Tin + tb + mi * 16 + lo) * NNP + node) * CINP;

    v8f accP[MT] = {}, accQ[MT] = {}, accR[MT] = {};
#pragma unroll
    for (int k0 = 0; k0 < KPAD; k0 += 32) {
        int kb = k0 + (hsel << 4);
        v16h vbP = *(const v16h*)(w0 + kb);
        v16h vbQ = *(const v16h*)(w1 + kb);
        v16h vbR = *(const v16h*)(w2 + kb);
        int ka0 = k0 + (hsel << 4);
        int ka1 = ka0 + 8;
        int dt0 = ka0 / CINP, ci0 = ka0 - dt0 * CINP;
        int dt1 = ka1 / CINP, ci1 = ka1 - dt1 * CINP;
        size_t o0 = (size_t)dt0 * (NNP * CINP) + ci0;
        size_t o1 = (size_t)dt1 * (NNP * CINP) + ci1;
#pragma unroll
        for (int mi = 0; mi < MT; ++mi) {
            v16h va = cat8(*(const v8h*)(x + abase[mi] + o0),
                           *(const v8h*)(x + abase[mi] + o1));
            accP[mi] = wmma_f16(va, vbP, accP[mi]);
            accQ[mi] = wmma_f16(va, vbQ, accQ[mi]);
            accR[mi] = wmma_f16(va, vbR, accR[mi]);
        }
    }
    float bP = bias[co], bQ = bias[COUT + co], bR = bias[2 * COUT + co];
    float scv = 1.f;
    if (BNIN) {
        scv = bnsc[node];
        const float ofv = bnof[node];
        bP += ofv * wsum[co];
        bQ += ofv * wsum[COUT + co];
        bR += ofv * wsum[2 * COUT + co];
    }
#pragma unroll
    for (int mi = 0; mi < MT; ++mi) {
#pragma unroll
        for (int r = 0; r < 8; ++r) {
            int tt = tb + mi * 16 + r + (hsel << 3);
            if (tt < Tout) {
                float p = accP[mi][r], q = accQ[mi][r], rr = accR[mi][r];
                if (BNIN) { p *= scv; q *= scv; rr *= scv; }
                p += bP; q += bQ; rr += bR;
                float v = p * (1.f / (1.f + __expf(-q))) + rr;
                out[((size_t)(b * Tout + tt) * NNP + node) * COUT + co] =
                    (_Float16)fmaxf(v, 0.f);
            }
        }
    }
}

// ---------------- L-hat propagation: out = alpha*L.z + beta*y --------------
// Lh: (224,224) f16; z,y,out: (BT,NNP,64) f16.  One block per (bt, m-pair):
// z[bt] staged transposed in LDS once; 8 waves consume via ds_load_b128.
template <bool DOBETA>
__global__ __launch_bounds__(256)
void k_lhat(const _Float16* __restrict__ Lh, const _Float16* __restrict__ z,
            const _Float16* __restrict__ y, _Float16* __restrict__ out,
            float alpha, float beta) {
    __shared__ _Float16 zt[64 * ZSTR];                  // ~29.7 KB

    const int mPair = blockIdx.x % 7;                   // 14 m-tiles / 2
    const int bt    = blockIdx.x / 7;

    {   // stage: z slice (224 k-rows x 64 n-cols, n contiguous) -> zt[n][k]
        const _Float16* zs = z + (size_t)bt * NNP * 64;
        for (int g = threadIdx.x; g < 224 * 8; g += 256) {
            int n8 = g & 7;
            int k  = g >> 3;
            v8h v = *(const v8h*)(zs + (size_t)k * 64 + n8 * 8);
#pragma unroll
            for (int j = 0; j < 8; ++j)
                zt[(n8 * 8 + j) * ZSTR + k] = v[j];
        }
    }
    __syncthreads();

    const int w    = threadIdx.x >> 5;
    const int mIdx = mPair * 2 + (w >> 2);
    const int nIdx = w & 3;
    const int l    = threadIdx.x & 31;
    const int lo   = l & 15;
    const int hsel = l >> 4;
    const int row  = mIdx * 16 + lo;                    // A row
    const int n    = nIdx * 16 + lo;                    // B column

    const _Float16* ab   = Lh + (size_t)row * NNP;
    const _Float16* zrow = &zt[n * ZSTR];

    v8f acc = {};
#pragma unroll
    for (int k0 = 0; k0 < NNP; k0 += 32) {
        v16h va = cat8(*(const v8h*)(ab + k0 + (hsel << 4)),
                       *(const v8h*)(ab + k0 + (hsel << 4) + 8));
        int kb = k0 + (hsel << 4);
        v16h vb = cat8(*(const v8h*)(zrow + kb),
                       *(const v8h*)(zrow + kb + 8));
        acc = wmma_f16(va, vb, acc);
    }
    const size_t cb = (size_t)bt * NNP * 64 + n;
    _Float16 yv[8];
    if (DOBETA) {
#pragma unroll
        for (int r = 0; r < 8; ++r)
            yv[r] = y[cb + (size_t)(mIdx * 16 + r + (hsel << 3)) * 64];
    }
#pragma unroll
    for (int r = 0; r < 8; ++r) {
        int rr = mIdx * 16 + r + (hsel << 3);
        float v = alpha * acc[r];
        if (DOBETA) v += beta * (float)yv[r];
        out[cb + (size_t)rr * 64] = (_Float16)v;
    }
}

// ------- Chebyshev combine: relu(t0.W0 + t1.W1 + t2.W2 + bias) -------------
// t*: (M=BT*NNP, 64) f16; Wt: (3,64,64)[g][h] f16. MB=2 row-tiles per wave.
__global__ __launch_bounds__(256)
void k_cheb_combine(const _Float16* __restrict__ t0, const _Float16* __restrict__ t1,
                    const _Float16* __restrict__ t2, const _Float16* __restrict__ Wt,
                    const float* __restrict__ bias, _Float16* __restrict__ out,
                    int nTiles) {
    int wave = blockIdx.x * 8 + (threadIdx.x >> 5);
    if (wave >= nTiles) return;
    int nIdx  = wave % 4;
    int mPair = wave / 4;
    const int l    = threadIdx.x & 31;
    const int lo   = l & 15;
    const int hsel = l >> 4;
    const int row0 = mPair * 32 + lo;
    const int n    = nIdx * 16 + lo;

    const _Float16* srcs[3] = { t0, t1, t2 };
    v8f acc0 = {}, acc1 = {};
#pragma unroll
    for (int s = 0; s < 3; ++s) {
        const _Float16* ab = srcs[s] + (size_t)row0 * 64;
        const _Float16* wb = Wt + ((size_t)s * 64 + n) * 64;
#pragma unroll
        for (int k0 = 0; k0 < 64; k0 += 32) {
            v16h vb = cat8(*(const v8h*)(wb + k0 + (hsel << 3)),
                           *(const v8h*)(wb + k0 + 16 + (hsel << 3)));
            v16h va0 = cat8(*(const v8h*)(ab + k0 + (hsel << 3)),
                            *(const v8h*)(ab + k0 + 16 + (hsel << 3)));
            v16h va1 = cat8(*(const v8h*)(ab + 16 * 64 + k0 + (hsel << 3)),
                            *(const v8h*)(ab + 16 * 64 + k0 + 16 + (hsel << 3)));
            acc0 = wmma_f16(va0, vb, acc0);
            acc1 = wmma_f16(va1, vb, acc1);
        }
    }
    const float bs = bias[n];
#pragma unroll
    for (int r = 0; r < 8; ++r) {
        int rr = mPair * 32 + r + (hsel << 3);
        out[(size_t)rr * 64 + n]        = (_Float16)fmaxf(acc0[r] + bs, 0.f);
        out[(size_t)(rr + 16) * 64 + n] = (_Float16)fmaxf(acc1[r] + bs, 0.f);
    }
}

// ------------- batchnorm statistics per node (no normalize pass) -----------
// sc = g/sqrt(var+eps), of = b - mean*sc; consumer applies them linearly.
template <int C>
__global__ __launch_bounds__(256)
void k_bn_stats(const _Float16* __restrict__ x, const float* __restrict__ g,
                const float* __restrict__ bsh, float* __restrict__ sc,
                float* __restrict__ of, int B, int T) {
    const int nd = blockIdx.x;
    const size_t cnt = (size_t)B * T * C;
    constexpr int CG = C / 8;
    float s = 0.f, s2 = 0.f;
    for (size_t jg = threadIdx.x; jg < cnt / 8; jg += blockDim.x) {
        size_t bt = jg / CG;  int cg = (int)(jg - bt * CG);
        v8h v = *(const v8h*)(x + (bt * NNP + nd) * C + cg * 8);
#pragma unroll
        for (int j = 0; j < 8; ++j) { float f = (float)v[j]; s += f; s2 += f * f; }
    }
    __shared__ float sh1[256], sh2[256];
    sh1[threadIdx.x] = s; sh2[threadIdx.x] = s2;
    __syncthreads();
    for (int st = 128; st > 0; st >>= 1) {
        if (threadIdx.x < (unsigned)st) {
            sh1[threadIdx.x] += sh1[threadIdx.x + st];
            sh2[threadIdx.x] += sh2[threadIdx.x + st];
        }
        __syncthreads();
    }
    if (threadIdx.x == 0) {
        const float inv = 1.f / (float)cnt;
        const float mean = sh1[0] * inv;
        const float var  = sh2[0] * inv - mean * mean;
        const float scv = g[nd] * rsqrtf(var + 1e-5f);
        sc[nd] = scv;
        of[nd] = bsh[nd] - mean * scv;
    }
}

// ------------------- final FC (16->3) with fused BN ------------------------
__global__ void k_fc(const _Float16* __restrict__ in, const float* __restrict__ fw,
                     const float* __restrict__ fb, const float* __restrict__ sc,
                     const float* __restrict__ of, float* __restrict__ out, int M) {
    int i = blockIdx.x * blockDim.x + threadIdx.x;
    if (i < M * 3) {
        int m = i / 3, gch = i - m * 3;
        int bt = m / NN, node = m - bt * NN;
        const _Float16* p = in + ((size_t)bt * NNP + node) * 16;
        float dot = 0.f, ws = 0.f;
#pragma unroll
        for (int k = 0; k < 16; ++k) {
            float w = fw[k * 3 + gch];
            dot += (float)p[k] * w;
            ws  += w;
        }
        out[i] = sc[node] * dot + of[node] * ws + fb[gch];
    }
}

// ---------------------------------------------------------------------------

extern "C" void kernel_launch(void* const* d_in, const int* in_sizes, int n_in,
                              void* d_out, int out_size, void* d_ws, size_t ws_size,
                              hipStream_t stream) {
    const float* x     = (const float*)d_in[0];
    const int*   ei    = (const int*)  d_in[1];
    const float* ea    = (const float*)d_in[2];
    const float* b1t1w = (const float*)d_in[3];
    const float* b1t1b = (const float*)d_in[4];
    const float* b1cw  = (const float*)d_in[5];
    const float* b1cb  = (const float*)d_in[6];
    const float* b1t2w = (const float*)d_in[7];
    const float* b1t2b = (const float*)d_in[8];
    const float* b1bng = (const float*)d_in[9];
    const float* b1bnb = (const float*)d_in[10];
    const float* b2t1w = (const float*)d_in[11];
    const float* b2t1b = (const float*)d_in[12];
    const float* b2cw  = (const float*)d_in[13];
    const float* b2cb  = (const float*)d_in[14];
    const float* b2t2w = (const float*)d_in[15];
    const float* b2t2b = (const float*)d_in[16];
    const float* b2bng = (const float*)d_in[17];
    const float* b2bnb = (const float*)d_in[18];
    const float* fcw   = (const float*)d_in[19];
    const float* fcb   = (const float*)d_in[20];

    // ---- workspace carve (~315 MB) ----
    char* ws = (char*)d_ws;
    size_t off = 0;
    auto carve = [&](size_t bytes) -> void* {
        void* p = ws + off;
        off = (off + bytes + 255) & ~(size_t)255;
        return p;
    };
    const size_t TAIL = 262144;                       // overshoot pad (halves)
    float*    Lf   = (float*)   carve((size_t)NN * NN * 4);
    float*    deg  = (float*)   carve((size_t)NN * 4);
    float*    sc1  = (float*)   carve((size_t)NN * 4);
    float*    of1  = (float*)   carve((size_t)NN * 4);
    float*    sc2  = (float*)   carve((size_t)NN * 4);
    float*    of2  = (float*)   carve((size_t)NN * 4);
    float*    ws3  = (float*)   carve((size_t)3 * 64 * 4);   // tap sums for wp3
    _Float16* Lh   = (_Float16*)carve((size_t)NNP * NNP * 2);
    _Float16* xp   = (_Float16*)carve(((size_t)32 * 64 * NNP * 16 + TAIL) * 2);
    _Float16* wp1  = (_Float16*)carve((size_t)3 * 64 * 160 * 2);
    _Float16* wp2  = (_Float16*)carve((size_t)3 * 128 * 576 * 2);
    _Float16* wp3  = (_Float16*)carve((size_t)3 * 64 * 640 * 2);
    _Float16* wp4  = (_Float16*)carve((size_t)3 * 16 * 320 * 2);
    _Float16* cw1  = (_Float16*)carve((size_t)3 * 64 * 64 * 2);
    _Float16* cw2  = (_Float16*)carve((size_t)3 * 64 * 64 * 2);
    const size_t ABUF = (size_t)1792 * NNP * 64;      // (32,56,224,64)
    _Float16* bufA = (_Float16*)carve((ABUF + TAIL) * 2);
    _Float16* bufB = (_Float16*)carve((ABUF + TAIL) * 2);
    _Float16* bufC = (_Float16*)carve((ABUF + TAIL) * 2);
    _Float16* bufD = (_Float16*)carve((ABUF + TAIL) * 2);
    _Float16* bufE = (_Float16*)carve(((size_t)1536 * NNP * 128 + TAIL) * 2);
    (void)ws_size; (void)in_sizes; (void)n_in; (void)out_size;

    // ---- preprocessing: L-hat + packed/padded f16 operands ----
    k_zero<<<(NN * NN + 255) / 256, 256, 0, stream>>>(Lf, NN * NN);
    k_zero<<<1, 256, 0, stream>>>(deg, NN);
    k_deg<<<(NE + 255) / 256, 256, 0, stream>>>(ei, ea, deg);
    k_build_L<<<(NE + 255) / 256, 256, 0, stream>>>(ei, ea, deg, Lf);
    k_pack_L<<<(NNP * NNP + 255) / 256, 256, 0, stream>>>(Lf, Lh);
    {
        int tot = 32 * 64 * NNP * 16;
        k_pack_x<<<(tot + 255) / 256, 256, 0, stream>>>(x, xp, tot);
    }
    k_pack_wt<<<(3 * 64 * 160 + 255) / 256, 256, 0, stream>>>(b1t1w, wp1, 9, 14, 16, 64, 160);
    k_pack_wt<<<(3 * 128 * 576 + 255) / 256, 256, 0, stream>>>(b1t2w, wp2, 9, 64, 64, 128, 576);
    k_pack_wt<<<(3 * 64 * 640 + 255) / 256, 256, 0, stream>>>(b2t1w, wp3, 5, 128, 128, 64, 640);
    k_pack_wt<<<(3 * 16 * 320 + 255) / 256, 256, 0, stream>>>(b2t2w, wp4, 5, 64, 64, 16, 320);
    k_wsum<<<1, 256, 0, stream>>>(wp3, ws3, 64, 640);
    k_pack_cw<<<(3 * 64 * 64 + 255) / 256, 256, 0, stream>>>(b1cw, cw1);
    k_pack_cw<<<(3 * 64 * 64 + 255) / 256, 256, 0, stream>>>(b2cw, cw2);

    int tiles;
    // ================= block 1 (T: 64 -> 56 -> 48) =================
    tiles = 32 * NN * 2 * 4;                           // mChunks=2 (MT=2), NT=4
    k_tconv<160, 16, 64, 2, false><<<(tiles + 7) / 8, 256, 0, stream>>>(
        xp, wp1, b1t1b, nullptr, nullptr, nullptr, bufA, 32, 64, 56, 2, tiles);

    k_lhat<false><<<1792 * 7, 256, 0, stream>>>(Lh, bufA, bufA, bufB, 1.f,  0.f);
    k_lhat<true ><<<1792 * 7, 256, 0, stream>>>(Lh, bufB, bufA, bufC, 2.f, -1.f);

    tiles = 1792 * 7 * 4;                              // M = 1792*224 rows, MB=2
    k_cheb_combine<<<(tiles + 7) / 8, 256, 0, stream>>>(
        bufA, bufB, bufC, cw1, b1cb, bufD, tiles);

    tiles = 32 * NN * 1 * 8;                           // mChunks=1 (MT=3), NT=8
    k_tconv<576, 64, 128, 3, false><<<(tiles + 7) / 8, 256, 0, stream>>>(
        bufD, wp2, b1t2b, nullptr, nullptr, nullptr, bufE, 32, 56, 48, 1, tiles);

    k_bn_stats<128><<<NN, 256, 0, stream>>>(bufE, b1bng, b1bnb, sc1, of1, 32, 48);

    // ================= block 2 (T: 48 -> 44 -> 40) =================
    tiles = 32 * NN * 1 * 4;                           // mChunks=1 (MT=3), NT=4
    k_tconv<640, 128, 64, 3, true><<<(tiles + 7) / 8, 256, 0, stream>>>(
        bufE, wp3, b2t1b, sc1, of1, ws3, bufA, 32, 48, 44, 1, tiles);

    k_lhat<false><<<1408 * 7, 256, 0, stream>>>(Lh, bufA, bufA, bufB, 1.f,  0.f);
    k_lhat<true ><<<1408 * 7, 256, 0, stream>>>(Lh, bufB, bufA, bufC, 2.f, -1.f);

    tiles = 1408 * 7 * 4;                              // M = 1408*224 rows, MB=2
    k_cheb_combine<<<(tiles + 7) / 8, 256, 0, stream>>>(
        bufA, bufB, bufC, cw2, b2cb, bufD, tiles);

    tiles = 32 * NN * 1 * 1;                           // mChunks=1 (MT=3), NT=1
    k_tconv<320, 64, 16, 3, false><<<(tiles + 7) / 8, 256, 0, stream>>>(
        bufD, wp4, b2t2b, nullptr, nullptr, nullptr, bufE, 32, 44, 40, 1, tiles);

    k_bn_stats<16><<<NN, 256, 0, stream>>>(bufE, b2bng, b2bnb, sc2, of2, 32, 40);

    // ================= FC head (BN2 fused) =================
    k_fc<<<(268800 * 3 + 255) / 256, 256, 0, stream>>>(
        bufE, fcw, fcb, sc2, of2, (float*)d_out, 268800);
}
